// GATLayer_44770739093962
// MI455X (gfx1250) — matrix operand order repaired and checked by reading, not compiled, not run on hardware
//
#include <hip/hip_runtime.h>
#include <hip/hip_bf16.h>
#include <math.h>

#define IN_DIM  128
#define HEADS   4
#define OUT_DIM 32
#define HD      128          // HEADS*OUT_DIM
#define LALPHA  0.2f
#define GEPS    1e-8f

typedef __attribute__((ext_vector_type(16))) __bf16 v16bf;
typedef __attribute__((ext_vector_type(8)))  float  v8f;

// ---------------------------------------------------------------------------
// K0: zero softmax denominator, init global max to -inf
// ---------------------------------------------------------------------------
__global__ void gat_init(float* __restrict__ denom, float* __restrict__ maxv, int n4) {
    int i = blockIdx.x * blockDim.x + threadIdx.x;
    if (i < n4) denom[i] = 0.0f;
    if (i == 0) *maxv = -INFINITY;
}

// ---------------------------------------------------------------------------
// K1: H = x @ W  and  R(=d_out) = x @ P, via v_wmma_f32_16x16x32_bf16.
// Block = 256 threads = 8 waves. Block handles 16 rows; wave w handles
// output columns [16w, 16w+16) of both W and P products. K=128 -> 4 WMMAs.
// ---------------------------------------------------------------------------
__device__ __forceinline__ void wmma_tile_store(
    const v16bf a[4], const float* __restrict__ B, float* __restrict__ O,
    int rowBase, int N, int colBase, int lane)
{
    const int col  = colBase + (lane & 15);
    const int ksel = lane >> 4;     // B: lanes 0-15 hold K 0-7,16-23, lanes 16-31 K 8-15,24-31
    v8f c = {};
#pragma unroll
    for (int kt = 0; kt < 4; ++kt) {
        const int kb = kt * 32 + ksel * 8;
        v16bf b;
#pragma unroll
        for (int i = 0; i < 16; ++i)
            b[i] = (__bf16)B[(size_t)(kb + (i < 8 ? i : i + 8)) * HD + col];
        c = __builtin_amdgcn_wmma_f32_16x16x32_bf16(
                false, a[kt], false, b, (short)0, c, false, false);
    }
    const int rtop = (lane >> 4) * 8;   // D: lanes 0-15 rows 0-7, lanes 16-31 rows 8-15
#pragma unroll
    for (int r = 0; r < 8; ++r) {
        int row = rowBase + rtop + r;
        if (row < N)
            O[(size_t)row * HD + colBase + (lane & 15)] = c[r];
    }
}

__global__ __launch_bounds__(256) void gat_gemm_wmma(
    const float* __restrict__ x, const float* __restrict__ W,
    const float* __restrict__ P, float* __restrict__ H,
    float* __restrict__ R, int N)
{
    const int wave = threadIdx.x >> 5;
    const int lane = threadIdx.x & 31;
    const int rowBase = blockIdx.x * 16;
    if (rowBase >= N) return;

    // A fragment: lane = row (mod 16); half selects K subrange within each 32-K tile
    const int arow = min(rowBase + (lane & 15), N - 1);
    const int half = lane >> 4;
    const float* xrow = x + (size_t)arow * IN_DIM;

    v16bf a[4];
#pragma unroll
    for (int kt = 0; kt < 4; ++kt) {
        const int k0 = kt * 32 + half * 8;
#pragma unroll
        for (int i = 0; i < 8; ++i) {
            a[kt][i]     = (__bf16)xrow[k0 + i];        // K = k0..k0+7
            a[kt][i + 8] = (__bf16)xrow[k0 + 16 + i];   // K = k0+16..k0+23
        }
    }

    wmma_tile_store(a, W, H, rowBase, N, wave * 16, lane);  // attention projection
    wmma_tile_store(a, P, R, rowBase, N, wave * 16, lane);  // residual projection
}

// ---------------------------------------------------------------------------
// K2: per-node attention scalars Hs = H.Aleft, Ht = H.Aright, stored (N,4)
// One wave per node; lane handles 4 dims; 8-lane groups = one head.
// ---------------------------------------------------------------------------
__global__ __launch_bounds__(256) void gat_hsht(
    const float* __restrict__ H, const float* __restrict__ AL,
    const float* __restrict__ AR, float* __restrict__ hs,
    float* __restrict__ ht, int N)
{
    const int wave = threadIdx.x >> 5;
    const int lane = threadIdx.x & 31;
    const int node = blockIdx.x * 8 + wave;
    if (node >= N) return;

    const float4 v  = *(const float4*)(H  + (size_t)node * HD + lane * 4);
    const float4 al = *(const float4*)(AL + lane * 4);   // (HEADS,OUT_DIM) flat == HD flat
    const float4 ar = *(const float4*)(AR + lane * 4);

    float dl = v.x*al.x + v.y*al.y + v.z*al.z + v.w*al.w;
    float dr = v.x*ar.x + v.y*ar.y + v.z*ar.z + v.w*ar.w;
#pragma unroll
    for (int off = 1; off < 8; off <<= 1) {
        dl += __shfl_xor(dl, off, 32);
        dr += __shfl_xor(dr, off, 32);
    }
    if ((lane & 7) == 0) {
        const int h = lane >> 3;
        hs[node * 4 + h] = dl;
        ht[node * 4 + h] = dr;
    }
}

// ---------------------------------------------------------------------------
// Edge-score helper (leaky relu of Hs[src]+Ht[tgt] per head)
// ---------------------------------------------------------------------------
__device__ __forceinline__ float leaky(float s) { return s >= 0.0f ? s : LALPHA * s; }

__device__ __forceinline__ void atomicMaxFloat(float* addr, float val) {
    if (val >= 0.0f) atomicMax((int*)addr, __float_as_int(val));
    else             atomicMin((unsigned int*)addr, (unsigned int)__float_as_int(val));
}

// ---------------------------------------------------------------------------
// K3: global max of edge scores
// ---------------------------------------------------------------------------
__global__ __launch_bounds__(256) void gat_max(
    const int* __restrict__ ei, const float* __restrict__ hs,
    const float* __restrict__ ht, float* __restrict__ maxv, int E)
{
    const int e = blockIdx.x * 256 + threadIdx.x;
    float m = -INFINITY;
    if (e < E) {
        const int s = ei[e], t = ei[E + e];
        const float4 a = *(const float4*)(hs + (size_t)s * 4);
        const float4 b = *(const float4*)(ht + (size_t)t * 4);
        m = fmaxf(fmaxf(leaky(a.x + b.x), leaky(a.y + b.y)),
                  fmaxf(leaky(a.z + b.z), leaky(a.w + b.w)));
    }
#pragma unroll
    for (int off = 1; off < 32; off <<= 1) m = fmaxf(m, __shfl_xor(m, off, 32));
    __shared__ float sm[8];
    if ((threadIdx.x & 31) == 0) sm[threadIdx.x >> 5] = m;
    __syncthreads();
    if (threadIdx.x < 8) {
        m = sm[threadIdx.x];
#pragma unroll
        for (int off = 1; off < 8; off <<= 1) m = fmaxf(m, __shfl_xor(m, off, 32));
        if (threadIdx.x == 0) atomicMaxFloat(maxv, m);
    }
}

// ---------------------------------------------------------------------------
// K4: softmax denominator: denom[src,h] += exp(s - M)
// ---------------------------------------------------------------------------
__global__ __launch_bounds__(256) void gat_denom(
    const int* __restrict__ ei, const float* __restrict__ hs,
    const float* __restrict__ ht, const float* __restrict__ maxv,
    float* __restrict__ denom, int E)
{
    const int e = blockIdx.x * 256 + threadIdx.x;
    if (e >= E) return;
    const float M = *maxv;
    const int s = ei[e], t = ei[E + e];
    const float4 a = *(const float4*)(hs + (size_t)s * 4);
    const float4 b = *(const float4*)(ht + (size_t)t * 4);
    atomicAdd(&denom[s * 4 + 0], __expf(leaky(a.x + b.x) - M));
    atomicAdd(&denom[s * 4 + 1], __expf(leaky(a.y + b.y) - M));
    atomicAdd(&denom[s * 4 + 2], __expf(leaky(a.z + b.z) - M));
    atomicAdd(&denom[s * 4 + 3], __expf(leaky(a.w + b.w) - M));
}

// ---------------------------------------------------------------------------
// K5: scatter  out[src, h, :] += alpha * H[tgt, h, :]
// One wave per edge: lane covers 4 dims; coalesced gather + contiguous atomics
// (agg buffer is 51 MB -> resident in the 192 MB L2, atomics stay on-chip)
// ---------------------------------------------------------------------------
__global__ __launch_bounds__(256) void gat_scatter(
    const int* __restrict__ ei, const float* __restrict__ hs,
    const float* __restrict__ ht, const float* __restrict__ maxv,
    const float* __restrict__ denom, const float* __restrict__ H,
    float* __restrict__ out, int E)
{
    const int wave = threadIdx.x >> 5;
    const int lane = threadIdx.x & 31;
    const int e = blockIdx.x * 8 + wave;
    if (e >= E) return;
    const int s = ei[e], t = ei[E + e];
    const float M = *maxv;
    const int h = lane >> 3;
    const float sc    = leaky(hs[s * 4 + h] + ht[t * 4 + h]);
    const float alpha = __expf(sc - M) / (denom[s * 4 + h] + GEPS);

    const float4 v = *(const float4*)(H + (size_t)t * HD + lane * 4);
    float* o = out + (size_t)s * HD + lane * 4;
    atomicAdd(o + 0, alpha * v.x);
    atomicAdd(o + 1, alpha * v.y);
    atomicAdd(o + 2, alpha * v.z);
    atomicAdd(o + 3, alpha * v.w);
}

// ---------------------------------------------------------------------------
// K6: in-place ELU (alpha = 1)
// ---------------------------------------------------------------------------
__global__ __launch_bounds__(256) void gat_elu(float* __restrict__ out, long n) {
    long i = (long)blockIdx.x * blockDim.x + threadIdx.x;
    if (i < n) {
        float v = out[i];
        out[i] = v > 0.0f ? v : (__expf(v) - 1.0f);
    }
}

// ---------------------------------------------------------------------------
extern "C" void kernel_launch(void* const* d_in, const int* in_sizes, int n_in,
                              void* d_out, int out_size, void* d_ws, size_t ws_size,
                              hipStream_t stream) {
    const float* x  = (const float*)d_in[0];
    const int*   ei = (const int*)  d_in[1];   // (2,E) int32 (jax x64 disabled)
    const float* W  = (const float*)d_in[2];
    const float* AL = (const float*)d_in[3];
    const float* AR = (const float*)d_in[4];
    const float* P  = (const float*)d_in[5];
    float* out = (float*)d_out;

    const int N = in_sizes[0] / IN_DIM;
    const int E = in_sizes[1] / 2;

    // workspace layout (floats): H[N*128] | hs[N*4] | ht[N*4] | denom[N*4] | maxv[1]
    float* H     = (float*)d_ws;
    float* hs    = H     + (size_t)N * HD;
    float* ht    = hs    + (size_t)N * 4;
    float* denom = ht    + (size_t)N * 4;
    float* maxv  = denom + (size_t)N * 4;

    gat_init   <<<(N * 4 + 255) / 256, 256, 0, stream>>>(denom, maxv, N * 4);
    gat_gemm_wmma<<<(N + 15) / 16,     256, 0, stream>>>(x, W, P, H, out, N);
    gat_hsht   <<<(N + 7) / 8,         256, 0, stream>>>(H, AL, AR, hs, ht, N);
    gat_max    <<<(E + 255) / 256,     256, 0, stream>>>(ei, hs, ht, maxv, E);
    gat_denom  <<<(E + 255) / 256,     256, 0, stream>>>(ei, hs, ht, maxv, denom, E);
    gat_scatter<<<(E + 7) / 8,         256, 0, stream>>>(ei, hs, ht, maxv, denom, H, out, E);
    gat_elu    <<<(int)(((long)N * HD + 255) / 256), 256, 0, stream>>>(out, (long)N * HD);
}
